// ImplicitPolygonInjector_22488448762525
// MI455X (gfx1250) — matrix operand order repaired and checked by reading, not compiled, not run on hardware
//
#include <hip/hip_runtime.h>
#include <math.h>

typedef __attribute__((ext_vector_type(16))) __bf16 v16bf;
typedef __attribute__((ext_vector_type(2)))  __bf16 v2bf;
typedef __attribute__((ext_vector_type(8)))  float  v8f;

#define C_IN 128
#define NPIX 16384   // B*H*W = 4*64*64

// native f32 -> bf16 (compiler emits hardware cvt, ideally packed forms)
static __device__ __forceinline__ __bf16 f2bf(float f) { return (__bf16)f; }

static __device__ __forceinline__ unsigned int pack2bf(float a, float b) {
  v2bf p;
  p[0] = (__bf16)a;
  p[1] = (__bf16)b;
  return __builtin_bit_cast(unsigned int, p);
}

// ---------------------------------------------------------------------------
// Stage 1: per pixel compute h_f = f @ W1_f (WMMA, bf16 in / f32 acc) and the
// scalar gate. One block = one tile of 16 consecutive pixels (same b,h row).
// ---------------------------------------------------------------------------
__global__ __launch_bounds__(256)
void k_stage1(const float* __restrict__ feat, const float* __restrict__ gw1,
              const float* __restrict__ gw2, const float* __restrict__ gb2,
              const float* __restrict__ mlp_w1,
              float* __restrict__ hf_ws, float* __restrict__ gate_ws)
{
  __shared__ float fLds[C_IN * 17];     // f[k][M], padded stride 17
  __shared__ float gw1Lds[C_IN * 32];   // gate_w1
  __shared__ float gateAcc[16];

  const int t    = threadIdx.x;
  const int wave = t >> 5, lane = t & 31;
  const int p0   = blockIdx.x << 4;                 // first pixel of tile
  const int b = p0 >> 12, h = (p0 >> 6) & 63, w0 = p0 & 63;

  // Resident bf16 B-fragments of W1_f (= mlp_w1 rows 0..127). Wave w owns
  // output channels n = 16w .. 16w+15.
  const int n     = (wave << 4) + (lane & 15);
  const int khalf = lane >> 4;
  v16bf bfrag[4];
#pragma unroll
  for (int kc = 0; kc < 4; ++kc)
#pragma unroll
    for (int j = 0; j < 8; ++j) {
      int k = kc * 32 + khalf * 16 + 2 * j;
      bfrag[kc][2 * j]     = f2bf(mlp_w1[k * C_IN + n]);
      bfrag[kc][2 * j + 1] = f2bf(mlp_w1[(k + 1) * C_IN + n]);
    }

  // Coalesced load of the 16x128 f-tile: f[pix][c] = feat[b][c][h][w0+pix]
  const float* fb = feat + ((size_t)b * C_IN) * 4096 + h * 64 + w0;
#pragma unroll
  for (int it = 0; it < 8; ++it) {
    int c = it * 16 + (t >> 4), pix = t & 15;
    fLds[c * 17 + pix] = fb[(size_t)c * 4096 + pix];
  }
#pragma unroll
  for (int it = 0; it < 16; ++it)
    gw1Lds[it * 256 + t] = gw1[it * 256 + t];
  if (t < 16) gateAcc[t] = 0.0f;
  __syncthreads();

  // h_f tile: D(16x16) += A(16x32) x B(32x16), 4 K-chunks
  const int M = lane & 15;
  v8f acc = {};
#pragma unroll
  for (int kc = 0; kc < 4; ++kc) {
    v16bf a;
#pragma unroll
    for (int e = 0; e < 16; ++e) {
      int k = kc * 32 + (khalf << 4) + e;
      a[e] = f2bf(fLds[k * 17 + M]);
    }
    acc = __builtin_amdgcn_wmma_f32_16x16x32_bf16(false, a, false, bfrag[kc],
                                                  (short)0, acc, false, false);
  }
#pragma unroll
  for (int i = 0; i < 8; ++i) {
    int Mr = i + ((lane >= 16) ? 8 : 0);
    hf_ws[(size_t)(p0 + Mr) * C_IN + n] = acc[i];
  }

  // Gate: 16 threads per pixel, 2 hidden comps each, LDS float-atomic reduce
  {
    int pix = t & 15, cp = t >> 4;
    float g0 = 0.f, g1 = 0.f;
    for (int k = 0; k < C_IN; ++k) {
      float fv = fLds[k * 17 + pix];
      g0 += fv * gw1Lds[k * 32 + 2 * cp];
      g1 += fv * gw1Lds[k * 32 + 2 * cp + 1];
    }
    g0 = (g0 >= 0.f) ? g0 : 0.2f * g0;
    g1 = (g1 >= 0.f) ? g1 : 0.2f * g1;
    atomicAdd(&gateAcc[pix], g0 * gw2[2 * cp] + g1 * gw2[2 * cp + 1]);
  }
  __syncthreads();
  if (t < 16) {
    float x = gateAcc[t] + gb2[0];
    gate_ws[p0 + t] = 1.0f / (1.0f + __expf(-x));
  }
}

// ---------------------------------------------------------------------------
// Stage 2: per pixel, A(16x128) = relu(h_f + h_g[m] + b1) built ONCE in LDS in
// fragment layout (packed 2xbf16 u32 stores); 8 waves hold the 8 N-tiles of W2
// resident in VGPRs and each does 4 WMMAs. Epilogue: bias+gate, float4 stores.
// ---------------------------------------------------------------------------
__global__ __launch_bounds__(256)
void k_stage2(const float* __restrict__ hf_ws, const float* __restrict__ gate_ws,
              const float* __restrict__ mlp_w1, const float* __restrict__ b1,
              const float* __restrict__ w2, const float* __restrict__ b2,
              float* __restrict__ out)
{
  __shared__ float hgLds[C_IN * 17];                     // (h_g + b1)[k][m]
  __shared__ __align__(32) unsigned int aLds[4 * 32 * 8]; // A frags, 2xbf16/u32
  __shared__ float hfs[C_IN];
  __shared__ float gateS;

  const int t    = threadIdx.x;
  const int wave = t >> 5, lane = t & 31;

  // h_g[m][k] + b1[k], m = 4*sy+sx, coords = {-0.75,-0.25,0.25,0.75}
#pragma unroll
  for (int it = 0; it < 8; ++it) {
    int idx = it * 256 + t;
    int k = idx >> 4, m = idx & 15;
    float cy = -0.75f + 0.5f * (float)(m >> 2);
    float cx = -0.75f + 0.5f * (float)(m & 3);
    hgLds[k * 17 + m] =
        cx * mlp_w1[128 * C_IN + k] + cy * mlp_w1[129 * C_IN + k] + b1[k];
  }

  // Resident bf16 B-fragments of W2 (128x128): wave w -> channels 16w..16w+15
  const int n     = (wave << 4) + (lane & 15);
  const int khalf = lane >> 4;
  v16bf bfrag[4];
#pragma unroll
  for (int kc = 0; kc < 4; ++kc)
#pragma unroll
    for (int j = 0; j < 8; ++j) {
      int k = kc * 32 + khalf * 16 + 2 * j;
      bfrag[kc][2 * j]     = f2bf(w2[k * C_IN + n]);
      bfrag[kc][2 * j + 1] = f2bf(w2[(k + 1) * C_IN + n]);
    }
  const float b2n = b2[n];
  __syncthreads();

  for (int p = blockIdx.x; p < NPIX; p += gridDim.x) {
    if (t < C_IN) hfs[t] = hf_ws[(size_t)p * C_IN + t];
    if (t == 255) gateS = gate_ws[p];
    __syncthreads();

    // Build shared A in exact fragment layout; each thread owns 8 consecutive
    // bf16 elements = 4 packed u32 LDS stores. elem idx = ((kc*32)+lv)*16 + e
#pragma unroll
    for (int j = 0; j < 4; ++j) {
      int idx = (t << 3) + 2 * j;            // even element index
      int kc = idx >> 9;
      int lv = (idx >> 4) & 31;
      int e  = idx & 15;
      int Mb = lv & 15, hs = lv >> 4;
      int k = kc * 32 + (hs << 4) + e;
      float v0 = hfs[k]     + hgLds[k * 17 + Mb];
      float v1 = hfs[k + 1] + hgLds[(k + 1) * 17 + Mb];
      v0 = (v0 > 0.f) ? v0 : 0.f;
      v1 = (v1 > 0.f) ? v1 : 0.f;
      aLds[idx >> 1] = pack2bf(v0, v1);
    }
    __syncthreads();

    v8f acc = {};
#pragma unroll
    for (int kc = 0; kc < 4; ++kc) {
      v16bf a = *reinterpret_cast<const v16bf*>(&aLds[(((kc << 5) + lane) << 3)]);
      acc = __builtin_amdgcn_wmma_f32_16x16x32_bf16(false, a, false, bfrag[kc],
                                                    (short)0, acc, false, false);
    }

    // Epilogue: rows M=0..3 (sy=syA, sx=0..3) are acc[0..3] of this lane ->
    // one float4 along wo; M=4..7 -> second float4 at ho+1.
    const float g  = gateS;
    const int  pb = p >> 12, ph = (p >> 6) & 63, pw = p & 63;
    const int  syA = (lane < 16) ? 0 : 2;
    float* o = out + (((size_t)pb * C_IN + n) << 16)
                   + (size_t)(ph * 4 + syA) * 256 + pw * 4;
    float4 v0, v1;
    v0.x = (acc[0] + b2n) * g; v0.y = (acc[1] + b2n) * g;
    v0.z = (acc[2] + b2n) * g; v0.w = (acc[3] + b2n) * g;
    v1.x = (acc[4] + b2n) * g; v1.y = (acc[5] + b2n) * g;
    v1.z = (acc[6] + b2n) * g; v1.w = (acc[7] + b2n) * g;
    *reinterpret_cast<float4*>(o)       = v0;
    *reinterpret_cast<float4*>(o + 256) = v1;
    __syncthreads();   // protect hfs/aLds before next pixel
  }
}

extern "C" void kernel_launch(void* const* d_in, const int* in_sizes, int n_in,
                              void* d_out, int out_size, void* d_ws, size_t ws_size,
                              hipStream_t stream) {
  const float* feat = (const float*)d_in[0];   // (4,128,64,64)
  const float* gw1  = (const float*)d_in[1];   // (128,32)
  const float* gw2  = (const float*)d_in[2];   // (32,1)
  const float* gb2  = (const float*)d_in[3];   // (1,)
  const float* w1   = (const float*)d_in[4];   // (130,128)
  const float* b1   = (const float*)d_in[5];   // (128,)
  const float* w2   = (const float*)d_in[6];   // (128,128)
  const float* b2   = (const float*)d_in[7];   // (128,)
  float* out = (float*)d_out;                  // (4,128,256,256)

  float* hf_ws   = (float*)d_ws;                 // NPIX*128 f32 = 8 MB
  float* gate_ws = hf_ws + (size_t)NPIX * C_IN;  // NPIX f32

  k_stage1<<<NPIX / 16, 256, 0, stream>>>(feat, gw1, gw2, gb2, w1, hf_ws, gate_ws);
  k_stage2<<<512, 256, 0, stream>>>(hf_ws, gate_ws, w1, b1, w2, b2, out);
}
